// Model_58523224375704
// MI455X (gfx1250) — matrix operand order repaired and checked by reading, not compiled, not run on hardware
//
#include <hip/hip_runtime.h>
#include <math.h>

typedef __attribute__((ext_vector_type(16))) _Float16 v16h;
typedef __attribute__((ext_vector_type(8)))  float    v8f;
typedef int v4i __attribute__((vector_size(16)));

#define BB   16
#define NN   1024
#define DD   128
#define BN   (BB * NN)      /* 16384 */
#define NBLK 15
#define EPSV 1e-5f

#define SLAB_STRIDE 40                    /* halves per d-row in LDS slab (pad vs bank conflicts) */
#define SLAB_HALVES (DD * SLAB_STRIDE)    /* 5120 halves = 10 KB */
#define AGG_STRIDE  136                   /* padded row stride for phase-2 staging */

#define AS1 __attribute__((address_space(1)))
#define AS3 __attribute__((address_space(3)))

#if __has_builtin(__builtin_amdgcn_global_load_async_to_lds_b128)
#define HAS_ASYNC_LDS 1
#else
#define HAS_ASYNC_LDS 0
#endif

#if __has_builtin(__builtin_amdgcn_s_wait_asynccnt)
#define WAIT_ASYNC() __builtin_amdgcn_s_wait_asynccnt(0)
#else
#define WAIT_ASYNC() asm volatile("s_wait_asynccnt 0" ::: "memory")
#endif

/* ---------------- one-time converts ---------------- */

__global__ void cvtL_kernel(const float* __restrict__ L, _Float16* __restrict__ LH) {
    size_t i = ((size_t)blockIdx.x * 256 + threadIdx.x) * 4;   // 16M elements / 4
    float4 v = *reinterpret_cast<const float4*>(L + i);
    union { _Float16 h[4]; uint2 u; } o;
    o.h[0] = (_Float16)v.x; o.h[1] = (_Float16)v.y;
    o.h[2] = (_Float16)v.z; o.h[3] = (_Float16)v.w;
    *reinterpret_cast<uint2*>(LH + i) = o.u;
}

__global__ void cvtW_kernel(const float* __restrict__ W, _Float16* __restrict__ WT) {
    int tid = blockIdx.x * 256 + threadIdx.x;                  // 15*128*128
    if (tid >= NBLK * DD * DD) return;
    int k    = tid / (DD * DD);
    int r    = tid - k * DD * DD;
    int dout = r >> 7, din = r & 127;
    WT[tid] = (_Float16)W[k * DD * DD + din * DD + dout];      // WT[k][dout][din]
}

/* ---------------- conv1: x1 = in @ W1 + b1 ; x2 = 0 ---------------- */

__global__ void conv1_kernel(const float* __restrict__ in, const float* __restrict__ W1,
                             const float* __restrict__ b1, float* __restrict__ p0,
                             float* __restrict__ p1) {
    int tid = blockIdx.x * 256 + threadIdx.x;                  // BN*DD = 2M
    int bm = tid >> 7, d = tid & 127;
    float v = b1[d] + in[bm * 3 + 0] * W1[d]
                    + in[bm * 3 + 1] * W1[DD + d]
                    + in[bm * 3 + 2] * W1[2 * DD + d];
    p0[tid] = v;
    p1[tid] = 0.0f;
}

/* ------------- BN stats: one block per channel, deterministic tree ------------- */

__global__ void bn_stats_kernel(const float* __restrict__ x, const float* __restrict__ g,
                                const float* __restrict__ be, float* __restrict__ ss) {
    const int d = blockIdx.x, t = threadIdx.x;
    float s = 0.0f, sq = 0.0f;
    for (int i = t; i < BN; i += 256) {
        float v = x[(size_t)i * DD + d];
        s += v; sq += v * v;
    }
    __shared__ float bs[256], bq[256];
    bs[t] = s; bq[t] = sq; __syncthreads();
    for (int o = 128; o > 0; o >>= 1) {
        if (t < o) { bs[t] += bs[t + o]; bq[t] += bq[t + o]; }
        __syncthreads();
    }
    if (t == 0) {
        float mean = bs[0] * (1.0f / BN);
        float var  = bq[0] * (1.0f / BN) - mean * mean;
        float sc   = g[d] * rsqrtf(var + EPSV);
        ss[d]      = sc;
        ss[DD + d] = be[d] - mean * sc;
    }
}

/* ------- h = elu(bn(a)) written TRANSPOSED as f16: hT[d][b*N+m], LDS-tiled ------- */

__global__ void bn_elu_tr_kernel(const float* __restrict__ x, const float* __restrict__ ss,
                                 _Float16* __restrict__ hT) {
    __shared__ float tile[32][33];
    const int tx = threadIdx.x, ty = threadIdx.y;              // (32, 8)
    const int d0 = blockIdx.x * 32;                            // 4 blocks
    const int r0 = blockIdx.y * 32;                            // 512 blocks
    #pragma unroll
    for (int j = 0; j < 4; ++j) {
        int r = r0 + j * 8 + ty, d = d0 + tx;
        float v = x[(size_t)r * DD + d] * ss[d] + ss[DD + d];
        v = v > 0.0f ? v : (expf(v) - 1.0f);                   // ELU, alpha=1
        tile[tx][j * 8 + ty] = v;
    }
    __syncthreads();
    #pragma unroll
    for (int j = 0; j < 4; ++j) {
        int dl = j * 8 + ty;
        hT[(size_t)(d0 + dl) * BN + r0 + tx] = (_Float16)tile[dl][tx];
    }
}

/* ---- fused scan step: Agg = L@h (WMMA, LDS-staged B) -> Agg@Wk (WMMA) + bias + skip ---- */

__global__ void __launch_bounds__(128)
fused_step_kernel(const _Float16* __restrict__ LH, const _Float16* __restrict__ hT,
                  const _Float16* __restrict__ WTk, const float* __restrict__ biask,
                  float* __restrict__ ax /* reads x2_old, writes a_new in place */) {
    __shared__ _Float16 smem[2 * SLAB_HALVES];                 /* 20 KB, aliased in phase 2 */
    const int lane  = threadIdx.x & 31;
    const int wave  = threadIdx.x >> 5;
    const int m     = lane & 15;
    const int half8  = (lane >> 4) << 3;
    const int half16 = (lane >> 4) << 4;
    const int b     = blockIdx.y;
    const int row0  = (blockIdx.x * 4 + wave) * 16;
    const int tid   = threadIdx.x;

    /* h slab staging: thread t owns channel row d=t; copies 4x16B (32 k-values) per step */
    const _Float16* hsrc = hT + (size_t)tid * BN + (size_t)b * NN;
    auto stage = [&](int buf, int k0) {
        _Float16* dst = smem + buf * SLAB_HALVES + tid * SLAB_STRIDE;
        #pragma unroll
        for (int part = 0; part < 4; ++part) {
#if HAS_ASYNC_LDS
            _Float16* s = const_cast<_Float16*>(hsrc + k0 + part * 8);
            __builtin_amdgcn_global_load_async_to_lds_b128(
                (AS1 v4i*)s, (AS3 v4i*)(dst + part * 8), 0, 0);
#else
            *reinterpret_cast<uint4*>(dst + part * 8) =
                *reinterpret_cast<const uint4*>(hsrc + k0 + part * 8);
#endif
        }
    };

    /* ---- phase 1: Agg[row0:row0+16, 0:128] = L[b] @ h[b], K = 1024 ---- */
    v8f acc[8] = {};
    const _Float16* Arow = LH + ((size_t)(b * NN + row0 + m)) * NN;

    stage(0, 0);
    for (int k0 = 0; k0 < NN; k0 += 32) {
        const int cur = (k0 >> 5) & 1;
        WAIT_ASYNC();
        __syncthreads();                                       /* slab[cur] visible to all waves */
        if (k0 + 32 < NN) stage(cur ^ 1, k0 + 32);             /* overlap next DMA with compute */

        v16h af;                                               /* A: K 0-7 / 16-23 per half */
        {   const _Float16* p = Arow + k0 + half8;
            *reinterpret_cast<uint4*>(&af)       = *reinterpret_cast<const uint4*>(p);
            *(reinterpret_cast<uint4*>(&af) + 1) = *reinterpret_cast<const uint4*>(p + 16);
        }
        /* load all 8 B-frags into distinct regs so ds_loads overlap the WMMA chain */
        const _Float16* slab = smem + cur * SLAB_HALVES;
        v16h bf[8];
        #pragma unroll
        for (int j = 0; j < 8; ++j) {
            const _Float16* q = slab + (j * 16 + m) * SLAB_STRIDE + half8;
            *reinterpret_cast<uint4*>(&bf[j])       = *reinterpret_cast<const uint4*>(q);
            *(reinterpret_cast<uint4*>(&bf[j]) + 1) = *reinterpret_cast<const uint4*>(q + 16);
        }
        #pragma unroll
        for (int j = 0; j < 8; ++j)
            acc[j] = __builtin_amdgcn_wmma_f32_16x16x32_f16(false, af, false, bf[j],
                                                            (short)0, acc[j], false, false);
    }

    /* ---- stage Agg strip (f16) in LDS; smem reused after barrier ---- */
    __syncthreads();
    _Float16* my = smem + wave * (16 * AGG_STRIDE);
    #pragma unroll
    for (int j = 0; j < 8; ++j)
        #pragma unroll
        for (int i = 0; i < 8; ++i)
            my[(i + half8) * AGG_STRIDE + j * 16 + m] = (_Float16)acc[j][i];

    /* ---- phase 2: Out = Agg @ Wk, K = 128 ---- */
    v8f acc2[8] = {};
    #pragma unroll
    for (int kk = 0; kk < DD; kk += 32) {
        v16h af;
        {   const _Float16* p = my + m * AGG_STRIDE + kk + half8;
            *reinterpret_cast<uint4*>(&af)       = *reinterpret_cast<const uint4*>(p);
            *(reinterpret_cast<uint4*>(&af) + 1) = *reinterpret_cast<const uint4*>(p + 16);
        }
        v16h wf[8];
        #pragma unroll
        for (int j = 0; j < 8; ++j) {
            const _Float16* p = WTk + (j * 16 + m) * DD + kk + half8;
            *reinterpret_cast<uint4*>(&wf[j])       = *reinterpret_cast<const uint4*>(p);
            *(reinterpret_cast<uint4*>(&wf[j]) + 1) = *reinterpret_cast<const uint4*>(p + 16);
        }
        #pragma unroll
        for (int j = 0; j < 8; ++j)
            acc2[j] = __builtin_amdgcn_wmma_f32_16x16x32_f16(false, af, false, wf[j],
                                                             (short)0, acc2[j], false, false);
    }

    /* ---- epilogue: a_new = x2_old + Out + bias (in place; ping-pong on host) ---- */
    #pragma unroll
    for (int j = 0; j < 8; ++j) {
        const int d = j * 16 + m;
        const float bias = biask[d];
        #pragma unroll
        for (int i = 0; i < 8; ++i) {
            const size_t idx = ((size_t)(b * NN + row0 + i + half8)) * DD + d;
            ax[idx] += acc2[j][i] + bias;
        }
    }
}

/* ------- final: x = elu(bn1(x1)+bn2(x2)); pooled[b,d] = sum_n mask*x / sum mask ------- */

__global__ void pool_kernel(const float* __restrict__ x1, const float* __restrict__ x2,
                            const float* __restrict__ ss1, const float* __restrict__ ss2,
                            const float* __restrict__ mask, float* __restrict__ pooled) {
    const int b = blockIdx.x, d = threadIdx.x;                 // 16 x 128
    const float s1 = ss1[d], a1 = ss1[DD + d];
    const float s2 = ss2[d], a2 = ss2[DD + d];
    float acc = 0.0f, cnt = 0.0f;
    for (int n = 0; n < NN; ++n) {
        float mk = mask[b * NN + n];
        size_t idx = ((size_t)(b * NN + n)) * DD + d;
        float t = x1[idx] * s1 + a1 + x2[idx] * s2 + a2;
        t = t > 0.0f ? t : (expf(t) - 1.0f);
        acc += mk * t; cnt += mk;
    }
    pooled[b * DD + d] = acc / cnt;
}

/* out[b,:] = pooled[b,:] @ W2 + b2   (tiny 16x128x128) */
__global__ void out_kernel(const float* __restrict__ pooled, const float* __restrict__ W2,
                           const float* __restrict__ b2, float* __restrict__ out) {
    const int b = blockIdx.x, e = threadIdx.x;
    float acc = b2[e];
    for (int d = 0; d < DD; ++d) acc += pooled[b * DD + d] * W2[d * DD + e];
    out[b * DD + e] = acc;
}

/* ---------------- host launch ---------------- */

extern "C" void kernel_launch(void* const* d_in, const int* in_sizes, int n_in,
                              void* d_out, int out_size, void* d_ws, size_t ws_size,
                              hipStream_t stream) {
    const float* inputs = (const float*)d_in[0];
    const float* L      = (const float*)d_in[1];
    const float* mask   = (const float*)d_in[2];
    const float* W1     = (const float*)d_in[3];
    const float* b1     = (const float*)d_in[4];
    const float* Wb     = (const float*)d_in[5];
    const float* bb     = (const float*)d_in[6];
    const float* gb     = (const float*)d_in[7];
    const float* beb    = (const float*)d_in[8];
    const float* g1     = (const float*)d_in[9];
    const float* be1    = (const float*)d_in[10];
    const float* g2     = (const float*)d_in[11];
    const float* be2    = (const float*)d_in[12];
    const float* W2     = (const float*)d_in[13];
    const float* b2     = (const float*)d_in[14];

    char* ws = (char*)d_ws;
    _Float16* LH  = (_Float16*)ws;  ws += (size_t)BB * NN * NN * 2;   // 32 MB
    _Float16* hT  = (_Float16*)ws;  ws += (size_t)DD * BN * 2;        // 4 MB
    float*    P0  = (float*)ws;     ws += (size_t)BN * DD * 4;        // 8 MB
    float*    P1  = (float*)ws;     ws += (size_t)BN * DD * 4;        // 8 MB
    _Float16* WT  = (_Float16*)ws;  ws += (size_t)NBLK * DD * DD * 2; // 480 KB
    float*    ss1 = (float*)ws;     ws += 2 * DD * 4;
    float*    ss2 = (float*)ws;     ws += 2 * DD * 4;
    float*    pooled = (float*)ws;  ws += (size_t)BB * DD * 4;

    cvtL_kernel<<<(BB * NN * NN) / (256 * 4), 256, 0, stream>>>(L, LH);
    cvtW_kernel<<<(NBLK * DD * DD + 255) / 256, 256, 0, stream>>>(Wb, WT);
    conv1_kernel<<<(BN * DD) / 256, 256, 0, stream>>>(inputs, W1, b1, P0, P1);

    float* A  = P0;   // scan carry a
    float* X2 = P1;   // scan carry b (skip)
    for (int it = 0; it < NBLK; ++it) {
        bn_stats_kernel<<<DD, 256, 0, stream>>>(A, gb + it * DD, beb + it * DD, ss1);
        bn_elu_tr_kernel<<<dim3(DD / 32, BN / 32), dim3(32, 8), 0, stream>>>(A, ss1, hT);
        fused_step_kernel<<<dim3(NN / 64, BB), 128, 0, stream>>>(
            LH, hT, WT + (size_t)it * DD * DD, bb + it * DD, X2);
        float* t = A; A = X2; X2 = t;                 // (a, b) <- (b + h, a)
    }

    bn_stats_kernel<<<DD, 256, 0, stream>>>(A,  g1, be1, ss1);
    bn_stats_kernel<<<DD, 256, 0, stream>>>(X2, g2, be2, ss2);
    pool_kernel<<<BB, DD, 0, stream>>>(A, X2, ss1, ss2, mask, pooled);
    out_kernel<<<BB, DD, 0, stream>>>(pooled, W2, b2, (float*)d_out);
}
